// DyNeXt_32169305047205
// MI455X (gfx1250) — matrix operand order repaired and checked by reading, not compiled, not run on hardware
//
#include <hip/hip_runtime.h>
#include <math.h>

// DyNeXt block for MI455X (gfx1250, wave32, WMMA), fragment-swizzled intermediates.
// pool -> fc/ln/fc (dyn 5x5 w) -> dwconv+chanLN (out1: B-frag bf16)
//      -> WMMA expand+GELU+sumsq (y: B-frag bf16, gx: f32 atomics)
//      -> GRN scale -> per-batch A-frag scale/swizzle of w_shrink
//      -> WMMA shrink + bias + residual.

#define B_   32
#define C_   128
#define H_   56
#define W_   56
#define HW_  3136
#define CR_  32
#define CE_  512
#define EPS_ 1e-6f

typedef __attribute__((ext_vector_type(16))) __bf16 v16bf;
typedef __attribute__((ext_vector_type(8)))  __bf16 v8bf;
typedef __attribute__((ext_vector_type(8)))  float  v8f;

// Fragment layouts (CDNA5 ISA 7.12.2, wave32), stored so each lane's fragment is
// 32 contiguous bytes at element offset (tile*32 + lane)*16:
//  A 16x32 tile (mt,kt): lane = 16*half + (row&15); element idx -> K:
//      k = kt*32 + 16*half + idx
//  B 32x16 tile (kt,nt): lane = 16*(kHigh) + (col&15); element idx -> k = kt*32+16*kHigh+idx

__device__ __forceinline__ float gelu_exact(float v) {
  return 0.5f * v * (1.0f + erff(v * 0.70710678118654752f));
}

// ---------- Stage 1: global average pool ----------
__global__ void dynext_pool(const float* __restrict__ x, float* __restrict__ pooled) {
  __shared__ float red[256];
  const int bc = blockIdx.x, t = threadIdx.x;
  const float* xp = x + (size_t)bc * HW_;
  float s = 0.f;
  for (int i = t; i < HW_; i += 256) s += xp[i];
  red[t] = s; __syncthreads();
  for (int o = 128; o > 0; o >>= 1) { if (t < o) red[t] += red[t + o]; __syncthreads(); }
  if (t == 0) pooled[bc] = red[0] * (1.0f / HW_);
}

// ---------- Stage 2: fc1 -> LN -> fc2 (dynamic conv weights) ----------
__global__ void dynext_fc_ln_fc(const float* __restrict__ pooled, const float* __restrict__ w_fc1,
                                const float* __restrict__ ln1w, const float* __restrict__ ln1b,
                                const float* __restrict__ w_fc2, float* __restrict__ dyn) {
  __shared__ float hdn[CR_], hn[CR_], stats[2];
  const int b = blockIdx.x, t = threadIdx.x;
  if (t < CR_) {
    float s = 0.f;
    for (int c = 0; c < C_; ++c) s += pooled[b * C_ + c] * w_fc1[t * C_ + c];
    hdn[t] = s;
  }
  __syncthreads();
  if (t == 0) {
    float s = 0.f, s2 = 0.f;
    for (int r = 0; r < CR_; ++r) { s += hdn[r]; s2 += hdn[r] * hdn[r]; }
    float u = s * (1.0f / CR_);
    stats[0] = u;
    stats[1] = rsqrtf(s2 * (1.0f / CR_) - u * u + EPS_);
  }
  __syncthreads();
  if (t < CR_) hn[t] = (hdn[t] - stats[0]) * stats[1] * ln1w[t] + ln1b[t];
  __syncthreads();
  for (int j = t; j < C_ * 25; j += 256) {
    float s = 0.f;
#pragma unroll
    for (int r = 0; r < CR_; ++r) s += hn[r] * w_fc2[j * CR_ + r];
    dyn[b * (C_ * 25) + j] = s;
  }
}

// ---------- Stage 3: dyn depthwise 5x5 + channel LN -> out1 in B-fragment layout ----------
__global__ void dynext_dwconv_chanln(const float* __restrict__ x, const float* __restrict__ dyn,
                                     const float* __restrict__ ln2w, const float* __restrict__ ln2b,
                                     __bf16* __restrict__ out1) {
  __shared__ float sconv[C_ * W_];   // [w*128 + c]
  __shared__ float swgt[C_ * 25];
  __shared__ float su[W_], srs[W_];
  const int b = blockIdx.x / H_;
  const int h = blockIdx.x % H_;
  const int t = threadIdx.x;
  for (int i = t; i < C_ * 25; i += 256) swgt[i] = dyn[b * (C_ * 25) + i];
  __syncthreads();
  for (int i = t; i < C_ * W_; i += 256) {
    const int w = i >> 7, c = i & 127;
    const float* xp = x + (size_t)(b * C_ + c) * HW_;
    const float* wp = swgt + c * 25;
    float acc = 0.f;
#pragma unroll
    for (int kh = 0; kh < 5; ++kh) {
      const int hh = h + kh - 2;
      if (hh < 0 || hh >= H_) continue;
#pragma unroll
      for (int kw = 0; kw < 5; ++kw) {
        const int ww = w + kw - 2;
        if (ww < 0 || ww >= W_) continue;
        acc += xp[hh * W_ + ww] * wp[kh * 5 + kw];
      }
    }
    sconv[i] = acc;
  }
  __syncthreads();
  if (t < W_) {
    float s = 0.f, s2 = 0.f;
    for (int c = 0; c < C_; ++c) { const float v = sconv[(t << 7) + c]; s += v; s2 += v * v; }
    const float u = s * (1.0f / C_);
    su[t] = u;
    srs[t] = rsqrtf(s2 * (1.0f / C_) - u * u + EPS_);
  }
  __syncthreads();
  // Write B-fragment layout (K=C rows, N=HW cols, KT=4): one v16bf (32B) per item.
  __bf16* out1b = out1 + (size_t)b * C_ * HW_;
  for (int item = t; item < W_ * 8; item += 256) {
    const int w = item >> 3, q = item & 7;
    const int kt = q >> 1, half = q & 1;
    const int c0 = kt * 32 + half * 16;
    const int p = h * W_ + w;
    const int nt = p >> 4, pi = p & 15;
    v16bf f;
#pragma unroll
    for (int j = 0; j < 16; ++j) {
      const int c = c0 + j;
      const float v = (sconv[(w << 7) + c] - su[w]) * srs[w] * ln2w[c] + ln2b[c];
      f[j] = (__bf16)v;
    }
    *(v16bf*)(out1b + ((size_t)(nt * 4 + kt) * 32 + (half << 4) + pi) * 16) = f;
  }
}

// ---------- w_expand -> bf16 A-fragment layout (32 mt x 4 kt) ----------
__global__ void dynext_swz_wexp(const float* __restrict__ w, __bf16* __restrict__ wsw) {
  const int tid = blockIdx.x * 256 + threadIdx.x;   // 32*4*32 = 4096
  if (tid >= 32 * 4 * 32) return;
  const int lane = tid & 31;
  const int kt = (tid >> 5) & 3;
  const int mt = tid >> 7;
  const int half = lane >> 4;
  const int row = mt * 16 + (lane & 15);
  v16bf f;
#pragma unroll
  for (int idx = 0; idx < 16; ++idx) {
    const int k = kt * 32 + 16 * half + idx;
    f[idx] = (__bf16)w[row * C_ + k];
  }
  *(v16bf*)(wsw + (size_t)tid * 16) = f;
}

// ---------- W_shrink @ grn_beta bias ----------
__global__ void dynext_wtb(const float* __restrict__ w_shrink, const float* __restrict__ beta,
                           float* __restrict__ wtb) {
  const int t = threadIdx.x;
  if (t < C_) {
    float s = 0.f;
    for (int o = 0; o < CE_; ++o) s += w_shrink[t * CE_ + o] * beta[o];
    wtb[t] = s;
  }
}

// ---------- Stage 4: WMMA expand (K=128) + GELU + sumsq; y in B-fragment layout (KT=16) ----------
__global__ void dynext_expand_gelu(const __bf16* __restrict__ out1, const __bf16* __restrict__ weSw,
                                   __bf16* __restrict__ y, float* __restrict__ gx) {
  const int lane  = threadIdx.x & 31;
  const int wid   = (blockIdx.x << 3) + (threadIdx.x >> 5);
  const int b     = wid / (32 * 49);
  const int rem   = wid % (32 * 49);
  const int mt    = rem / 49;
  const int strip = rem % 49;
  const __bf16* Bb = out1 + (size_t)b * C_ * HW_;

  const v8f vz = {0.f, 0.f, 0.f, 0.f, 0.f, 0.f, 0.f, 0.f};
  v8f acc[4] = {vz, vz, vz, vz};
#pragma unroll
  for (int kt = 0; kt < 4; ++kt) {
    const v16bf a = *(const v16bf*)(weSw + ((size_t)(mt * 4 + kt) * 32 + lane) * 16);
    v16bf bfr[4];
#pragma unroll
    for (int n = 0; n < 4; ++n) {
      const int nt = strip * 4 + n;
      bfr[n] = *(const v16bf*)(Bb + ((size_t)(nt * 4 + kt) * 32 + lane) * 16);
    }
#pragma unroll
    for (int n = 0; n < 4; ++n)
      acc[n] = __builtin_amdgcn_wmma_f32_16x16x32_bf16(false, a, false, bfr[n],
                                                       (short)0, acc[n], false, false);
  }
  const int half8 = ((lane >> 4) & 1) << 3;
  const int col   = lane & 15;
  float part[8];
#pragma unroll
  for (int r = 0; r < 8; ++r) part[r] = 0.f;
  // y B-fragment coords for output rows o = mt*16 + r + half8:
  //   kt_o = mt>>1, kHigh = mt&1, j = r + half8 (contiguous over r) -> one 16B store per tile.
  __bf16* yb = y + (size_t)b * CE_ * HW_;
  const int kt_o  = mt >> 1;
  const int lanep = ((mt & 1) << 4) + col;
#pragma unroll
  for (int n = 0; n < 4; ++n) {
    const int nt = strip * 4 + n;
    v8bf pk;
#pragma unroll
    for (int r = 0; r < 8; ++r) {
      const float g = gelu_exact(acc[n][r]);
      part[r] += g * g;
      pk[r] = (__bf16)g;
    }
    *(v8bf*)(yb + ((size_t)(nt * 16 + kt_o) * 32 + lanep) * 16 + half8) = pk;
  }
#pragma unroll
  for (int r = 0; r < 8; ++r) {
    float s = part[r];
    s += __shfl_xor(s, 1);
    s += __shfl_xor(s, 2);
    s += __shfl_xor(s, 4);
    s += __shfl_xor(s, 8);
    if ((lane & 15) == 0) atomicAdd(gx + b * CE_ + mt * 16 + r + half8, s);
  }
}

// ---------- Stage 5: GRN scale s[b,o] = gamma*nx + 1 ----------
__global__ void dynext_grn_scale(const float* __restrict__ gx, const float* __restrict__ gamma,
                                 float* __restrict__ s) {
  __shared__ float red[256];
  const int b = blockIdx.x, t = threadIdx.x;
  const float g0 = sqrtf(gx[b * CE_ + t]);
  const float g1 = sqrtf(gx[b * CE_ + t + 256]);
  red[t] = g0 + g1; __syncthreads();
  for (int o = 128; o > 0; o >>= 1) { if (t < o) red[t] += red[t + o]; __syncthreads(); }
  const float mean = red[0] * (1.0f / CE_);
  s[b * CE_ + t]       = gamma[t]       * (g0 / (mean + EPS_)) + 1.0f;
  s[b * CE_ + t + 256] = gamma[t + 256] * (g1 / (mean + EPS_)) + 1.0f;
}

// ---------- per-batch scaled w_shrink -> bf16 A-fragment layout (b x 8 mt x 16 kt) ----------
__global__ void dynext_shrink_prep(const float* __restrict__ wshr, const float* __restrict__ sscale,
                                   __bf16* __restrict__ wsw) {
  const int tid = blockIdx.x * 256 + threadIdx.x;  // 32*8*16*32 = 131072
  const int lane = tid & 31;
  const int kt = (tid >> 5) & 15;
  const int mt = (tid >> 9) & 7;
  const int b  = tid >> 12;
  const int half = lane >> 4;
  const int row = mt * 16 + (lane & 15);
  const float* sb = sscale + b * CE_;
  v16bf f;
#pragma unroll
  for (int idx = 0; idx < 16; ++idx) {
    const int k = kt * 32 + 16 * half + idx;
    f[idx] = (__bf16)(wshr[row * CE_ + k] * sb[k]);
  }
  *(v16bf*)(wsw + (size_t)tid * 16) = f;
}

// ---------- Stage 6: WMMA shrink (K=512) + bias + residual ----------
__global__ void dynext_shrink_out(const __bf16* __restrict__ y, const __bf16* __restrict__ wsSw,
                                  const float* __restrict__ wtb, const float* __restrict__ x,
                                  float* __restrict__ out) {
  const int lane  = threadIdx.x & 31;
  const int wid   = (blockIdx.x << 3) + (threadIdx.x >> 5);
  const int b     = wid / (8 * 49);
  const int rem   = wid % (8 * 49);
  const int mt    = rem / 49;
  const int strip = rem % 49;
  const __bf16* Bb = y + (size_t)b * CE_ * HW_;
  const __bf16* Ab = wsSw + (size_t)(b * 8 + mt) * 16 * 32 * 16;

  const v8f vz = {0.f, 0.f, 0.f, 0.f, 0.f, 0.f, 0.f, 0.f};
  v8f acc[4] = {vz, vz, vz, vz};
#pragma unroll 2
  for (int kt = 0; kt < 16; ++kt) {
    const v16bf a = *(const v16bf*)(Ab + ((size_t)kt * 32 + lane) * 16);
    v16bf bfr[4];
#pragma unroll
    for (int n = 0; n < 4; ++n) {
      const int nt = strip * 4 + n;
      bfr[n] = *(const v16bf*)(Bb + ((size_t)(nt * 16 + kt) * 32 + lane) * 16);
    }
#pragma unroll
    for (int n = 0; n < 4; ++n)
      acc[n] = __builtin_amdgcn_wmma_f32_16x16x32_bf16(false, a, false, bfr[n],
                                                       (short)0, acc[n], false, false);
  }
  const int half8 = ((lane >> 4) & 1) << 3;
  const int col   = lane & 15;
#pragma unroll
  for (int n = 0; n < 4; ++n) {
    const int p = (strip * 4 + n) * 16 + col;
#pragma unroll
    for (int r = 0; r < 8; ++r) {
      const int c = mt * 16 + r + half8;
      const size_t idx = (size_t)(b * C_ + c) * HW_ + p;
      out[idx] = acc[n][r] + wtb[c] + x[idx];
    }
  }
}

// ---------- host launcher ----------
extern "C" void kernel_launch(void* const* d_in, const int* in_sizes, int n_in,
                              void* d_out, int out_size, void* d_ws, size_t ws_size,
                              hipStream_t stream) {
  const float* x        = (const float*)d_in[0];
  const float* w_fc1    = (const float*)d_in[1];
  const float* ln1_w    = (const float*)d_in[2];
  const float* ln1_b    = (const float*)d_in[3];
  const float* w_fc2    = (const float*)d_in[4];
  const float* ln2_w    = (const float*)d_in[5];
  const float* ln2_b    = (const float*)d_in[6];
  const float* w_expand = (const float*)d_in[7];
  const float* w_shrink = (const float*)d_in[8];
  const float* grn_g    = (const float*)d_in[9];
  const float* grn_b    = (const float*)d_in[10];
  float* out = (float*)d_out;

  char* p = (char*)d_ws;
  float*  pooled = (float*)p;  p += (size_t)B_ * C_ * 4;                 // 16 KB
  float*  dyn    = (float*)p;  p += (size_t)B_ * C_ * 25 * 4;           // 400 KB
  float*  gx     = (float*)p;  p += (size_t)B_ * CE_ * 4;               // 64 KB
  float*  sscale = (float*)p;  p += (size_t)B_ * CE_ * 4;               // 64 KB
  float*  wtb    = (float*)p;  p += 512;                                // 128 f32 (padded)
  __bf16* weSw   = (__bf16*)p; p += (size_t)CE_ * C_ * 2;               // 128 KB (A-frag)
  __bf16* wsSw   = (__bf16*)p; p += (size_t)B_ * C_ * CE_ * 2;          // 4 MB  (A-frag, per-b scaled)
  __bf16* out1   = (__bf16*)p; p += (size_t)B_ * C_ * HW_ * 2;          // 25.7 MB (B-frag)
  __bf16* y      = (__bf16*)p; p += (size_t)B_ * CE_ * HW_ * 2;         // 102.8 MB (B-frag)
  (void)p; (void)ws_size; (void)in_sizes; (void)n_in; (void)out_size;

  hipMemsetAsync(gx, 0, (size_t)B_ * CE_ * 4, stream);

  dynext_pool<<<B_ * C_, 256, 0, stream>>>(x, pooled);
  dynext_fc_ln_fc<<<B_, 256, 0, stream>>>(pooled, w_fc1, ln1_w, ln1_b, w_fc2, dyn);
  dynext_dwconv_chanln<<<B_ * H_, 256, 0, stream>>>(x, dyn, ln2_w, ln2_b, out1);
  dynext_swz_wexp<<<16, 256, 0, stream>>>(w_expand, weSw);
  dynext_wtb<<<1, 128, 0, stream>>>(w_shrink, grn_b, wtb);
  // expand: 32 b * 32 mtiles * 49 strips = 50176 waves, 8 waves/block
  dynext_expand_gelu<<<(B_ * 32 * 49) / 8, 256, 0, stream>>>(out1, weSw, y, gx);
  dynext_grn_scale<<<B_, 256, 0, stream>>>(gx, grn_g, sscale);
  dynext_shrink_prep<<<512, 256, 0, stream>>>(w_shrink, sscale, wsSw);
  // shrink: 32 b * 8 mtiles * 49 strips = 12544 waves, 8 waves/block
  dynext_shrink_out<<<(B_ * 8 * 49) / 8, 256, 0, stream>>>(y, wsSw, wtb, x, out);
}
